// Latent_Learning_6640019440168
// MI455X (gfx1250) — matrix operand order repaired and checked by reading, not compiled, not run on hardware
//
#include <hip/hip_runtime.h>
#include <cstdint>

#define N_NODES 40000
#define N_REL   400
#define N_EDGES 100000
#define D_IN    200
#define D_OUT   200
#define NHEADS  2
#define ALPHA_S 0.2f
#define CAT_K   (D_IN + NHEADS * D_OUT)   // 600
#define OUT_W   (NHEADS * D_OUT)          // 400

typedef __attribute__((ext_vector_type(16))) _Float16 v16h;
typedef __attribute__((ext_vector_type(8)))  float    v8f;

// ---------------------------------------------------------------- helpers
__device__ __forceinline__ float wave_sum(float v) {
#pragma unroll
  for (int off = 16; off > 0; off >>= 1) v += __shfl_xor(v, off, 32);
  return v;
}

__device__ __forceinline__ void atomicMaxF(float* addr, float val) {
  if (val >= 0.f) atomicMax((int*)addr, __float_as_int(val));
  else            atomicMin((unsigned int*)addr, __float_as_uint(val));
}

// ---------------------------------------------------------------- tiny precompute
// u[h][j] = sum_k W_src[h][j][k] * a[h][k]      (first 800 threads)
// v[h][j] = sum_k W_dst[h][j][k] * a[h][200+k]  (next 800 threads)
__global__ void k_uv(const float* __restrict__ W_src, const float* __restrict__ W_dst,
                     const float* __restrict__ a_vec, float* __restrict__ u,
                     float* __restrict__ v) {
  int t = blockIdx.x * blockDim.x + threadIdx.x;
  if (t >= 2 * 4 * D_IN) return;
  int isV = (t >= 4 * D_IN);
  int t2 = t % (4 * D_IN);
  int h = t2 / D_IN, j = t2 % D_IN;
  const float* W = isV ? W_dst : W_src;
  const float* a = a_vec + h * (2 * D_OUT) + (isV ? D_OUT : 0);
  float s = 0.f;
  for (int k = 0; k < D_OUT; ++k)
    s += W[((size_t)h * D_IN + j) * D_OUT + k] * a[k];
  (isV ? v : u)[h * D_IN + j] = s;
}

// dlog[h][r] = relation_emb[r (+200 for head-heads)] . v[h]
__global__ void k_dlogit(const float* __restrict__ relemb, const float* __restrict__ v,
                         float* __restrict__ dlog) {
  int t = blockIdx.x * blockDim.x + threadIdx.x;
  if (t >= 4 * N_REL) return;
  int h = t / N_REL, r = t % N_REL;
  int row = (h < NHEADS) ? (r + N_REL / 2) : r;
  float s = 0.f;
  for (int j = 0; j < D_IN; ++j)
    s += relemb[(size_t)row * D_IN + j] * v[h * D_IN + j];
  dlog[t] = s;
}

__global__ void k_init(float* __restrict__ mbuf, float* __restrict__ denom,
                       float* __restrict__ agg) {
  int t = blockIdx.x * blockDim.x + threadIdx.x;
  if (t < 4 * N_REL)                         mbuf[t] = -INFINITY;
  else if (t < 8 * N_REL)                    denom[t - 4 * N_REL] = 0.f;
  else if (t < 8 * N_REL + 4 * N_REL * D_IN) agg[t - 8 * N_REL] = 0.f;
}

// ---------------------------------------------------------------- edge passes
// wave per (head,edge): logit = lrelu(src.u + dlog[h][et]); segment max
__global__ __launch_bounds__(256)
void k_edge_logits(const int* __restrict__ el, const int* __restrict__ et,
                   const float* __restrict__ ent, const float* __restrict__ u,
                   const float* __restrict__ dlog, float* __restrict__ mbuf,
                   float* __restrict__ ebuf) {
  int wave = blockIdx.x * blockDim.y + threadIdx.y;
  if (wave >= 4 * N_EDGES) return;
  int lane = threadIdx.x;
  int h = wave / N_EDGES, e = wave % N_EDGES;
  int src = el[(h < NHEADS ? 0 : 1) * N_EDGES + e];
  int r = et[e];
  float acc = 0.f;
  for (int j = lane; j < D_IN; j += 32)
    acc += ent[(size_t)src * D_IN + j] * u[h * D_IN + j];
  acc = wave_sum(acc);
  if (lane == 0) {
    float x = acc + dlog[h * N_REL + r];
    float lr = x > 0.f ? x : ALPHA_S * x;
    ebuf[wave] = lr;
    atomicMaxF(&mbuf[h * N_REL + r], lr);
  }
}

// thread per (head,edge): ex = exp(e - m); denom += ex
__global__ void k_edge_exp(const int* __restrict__ et, const float* __restrict__ mbuf,
                           float* __restrict__ ebuf, float* __restrict__ denom) {
  int t = blockIdx.x * blockDim.x + threadIdx.x;
  if (t >= 4 * N_EDGES) return;
  int h = t / N_EDGES, e = t % N_EDGES;
  int r = et[e];
  float ex = __expf(ebuf[t] - mbuf[h * N_REL + r]);
  ebuf[t] = ex;
  atomicAdd(&denom[h * N_REL + r], ex);
}

// wave per (head,edge): agg[h][et] += attn * src_emb
__global__ __launch_bounds__(256)
void k_edge_agg(const int* __restrict__ el, const int* __restrict__ et,
                const float* __restrict__ ent, const float* __restrict__ ebuf,
                const float* __restrict__ denom, float* __restrict__ agg) {
  int wave = blockIdx.x * blockDim.y + threadIdx.y;
  if (wave >= 4 * N_EDGES) return;
  int lane = threadIdx.x;
  int h = wave / N_EDGES, e = wave % N_EDGES;
  int src = el[(h < NHEADS ? 0 : 1) * N_EDGES + e];
  int r = et[e];
  float attn = ebuf[wave] / (denom[h * N_REL + r] + 1e-16f);
  for (int j = lane; j < D_IN; j += 32)
    atomicAdd(&agg[((size_t)h * N_REL + r) * D_IN + j],
              attn * ent[(size_t)src * D_IN + j]);
}

// ---------------------------------------------------------------- head projection
// wave per (r,c), c in [0,400): relcat[r][c] = elu(agg[ha][r] . Wsrc[ha][:,col])
//                                            + elu(agg[ha+2][r] . Wsrc[ha+2][:,col])
__global__ __launch_bounds__(256)
void k_head_out(const float* __restrict__ agg, const float* __restrict__ W_src,
                float* __restrict__ relcat) {
  int wave = blockIdx.x * blockDim.y + threadIdx.y;
  if (wave >= N_REL * OUT_W) return;
  int lane = threadIdx.x;
  int r = wave / OUT_W, c = wave % OUT_W;
  int ha = (c < D_OUT) ? 0 : 1;
  int col = c % D_OUT;
  float s0 = 0.f, s1 = 0.f;
  for (int j = lane; j < D_IN; j += 32) {
    float a0 = agg[((size_t)ha * N_REL + r) * D_IN + j];
    float a1 = agg[((size_t)(ha + 2) * N_REL + r) * D_IN + j];
    s0 += a0 * W_src[((size_t)ha * D_IN + j) * D_OUT + col];
    s1 += a1 * W_src[((size_t)(ha + 2) * D_IN + j) * D_OUT + col];
  }
  s0 = wave_sum(s0);
  s1 = wave_sum(s1);
  if (lane == 0) {
    float e0 = s0 > 0.f ? s0 : __expf(s0) - 1.f;
    float e1 = s1 > 0.f ? s1 : __expf(s1) - 1.f;
    relcat[(size_t)r * CAT_K + c] = e0 + e1;
  }
}

// relcat[r][400+j] = relation_emb[r][j]
__global__ void k_cat_rel(const float* __restrict__ relemb, float* __restrict__ relcat) {
  int t = blockIdx.x * blockDim.x + threadIdx.x;
  if (t >= N_REL * D_IN) return;
  int r = t / D_IN, j = t % D_IN;
  relcat[(size_t)r * CAT_K + OUT_W + j] = relemb[t];
}

// ---------------------------------------------------------------- WMMA GEMM
// C[M,N] = A[M,K] @ B[K,N], f32 in/out, f16 multiply + f32 accumulate.
// K / LDA / LDB / LDC are compile-time so every inner-loop access is an
// immediate offset off a hoisted base pointer (clause-able loads, no 64-bit
// address chains). Each wave computes an MT*16 x 16 strip (M % (MT*16)==0);
// the K-tail (K%32) is handled once with run-level predication.
template<int MT, int K, int LDA, int LDB, int LDC>
__global__ __launch_bounds__(128)
void k_wmma_gemm(const float* __restrict__ A, const float* __restrict__ B,
                 float* __restrict__ C, int M, int N) {
  const int wave    = blockIdx.x * blockDim.y + threadIdx.y;
  const int tilesN  = N >> 4;
  const int tilesMB = M / (MT * 16);
  if (wave >= tilesMB * tilesN) return;          // wave-uniform: EXEC stays full
  const int lane    = threadIdx.x;
  const int rowBase = (wave / tilesN) * (MT * 16);
  const int colBase = (wave % tilesN) << 4;

  const int m      = lane & 15;
  const int aKhalf = (lane < 16) ? 0 : 8;        // A 16-bit 16x32 layout
  const int n      = lane & 15;
  const int bKbase = (lane < 16) ? 0 : 8;        // B 16-bit 32x16 layout

  // hoisted base pointers: inner loop uses only constant offsets + induction
  const float* bp = B + colBase + n + bKbase * LDB;
  const float* ap[MT];
#pragma unroll
  for (int i = 0; i < MT; ++i)
    ap[i] = A + (size_t)(rowBase + i * 16 + m) * LDA + aKhalf;

  v8f acc[MT];
#pragma unroll
  for (int i = 0; i < MT; ++i) acc[i] = (v8f){0.f,0.f,0.f,0.f,0.f,0.f,0.f,0.f};

  constexpr int KF = K & ~31;
  for (int kk = 0; kk < KF; kk += 32) {
    v16h b;
#pragma unroll
    for (int i = 0; i < 16; ++i)
      b[i] = (_Float16)bp[kk * LDB + (i < 8 ? i : i + 8) * LDB];   // i*LDB: immediate offsets
#pragma unroll
    for (int i = 0; i < MT; ++i) {
      const float* ar = ap[i] + kk;
      const float4 a0 = *(const float4*)(ar);
      const float4 a1 = *(const float4*)(ar + 4);
      const float4 a2 = *(const float4*)(ar + 16);
      const float4 a3 = *(const float4*)(ar + 20);
      v16h a;
      a[0]  = (_Float16)a0.x;  a[1]  = (_Float16)a0.y;
      a[2]  = (_Float16)a0.z;  a[3]  = (_Float16)a0.w;
      a[4]  = (_Float16)a1.x;  a[5]  = (_Float16)a1.y;
      a[6]  = (_Float16)a1.z;  a[7]  = (_Float16)a1.w;
      a[8]  = (_Float16)a2.x;  a[9]  = (_Float16)a2.y;
      a[10] = (_Float16)a2.z;  a[11] = (_Float16)a2.w;
      a[12] = (_Float16)a3.x;  a[13] = (_Float16)a3.y;
      a[14] = (_Float16)a3.z;  a[15] = (_Float16)a3.w;
      acc[i] = __builtin_amdgcn_wmma_f32_16x16x32_f16(
          false, a, false, b, (short)0, acc[i], false, false);
    }
  }

  if constexpr (KF < K) {                        // one tail step (K % 32)
    v16h b;
#pragma unroll
    for (int i = 0; i < 16; ++i) {
      int kb = KF + bKbase + (i < 8 ? i : i + 8);
      b[i] = (kb < K) ? (_Float16)bp[(size_t)(KF + (i < 8 ? i : i + 8)) * LDB] : (_Float16)0.f;
    }
#pragma unroll
    for (int i = 0; i < MT; ++i) {
      v16h a;
#pragma unroll
      for (int run = 0; run < 2; ++run) {        // runs of 8 contiguous k's
        const int rs = KF + run * 16;            // + aKhalf already in ap[i]
        if (rs + aKhalf + 8 <= K) {              // whole run in-bounds
          const float4 q0 = *(const float4*)(ap[i] + rs);
          const float4 q1 = *(const float4*)(ap[i] + rs + 4);
          a[8*run + 0] = (_Float16)q0.x;  a[8*run + 1] = (_Float16)q0.y;
          a[8*run + 2] = (_Float16)q0.z;  a[8*run + 3] = (_Float16)q0.w;
          a[8*run + 4] = (_Float16)q1.x;  a[8*run + 5] = (_Float16)q1.y;
          a[8*run + 6] = (_Float16)q1.z;  a[8*run + 7] = (_Float16)q1.w;
        } else {                                 // partial/empty run
#pragma unroll
          for (int q = 0; q < 8; ++q)
            a[8*run + q] = (rs + aKhalf + q < K) ? (_Float16)ap[i][rs + q]
                                                 : (_Float16)0.f;
        }
      }
      acc[i] = __builtin_amdgcn_wmma_f32_16x16x32_f16(
          false, a, false, b, (short)0, acc[i], false, false);
    }
  }

  const int mb = (lane < 16) ? 0 : 8;            // C 16x16 f32 layout
  float* cp = C + (size_t)(rowBase + mb) * LDC + colBase + n;
#pragma unroll
  for (int i = 0; i < MT; ++i) {
#pragma unroll
    for (int r = 0; r < 8; ++r)
      cp[(i * 16 + r) * LDC] = acc[i][r];
  }
}

// ---------------------------------------------------------------- launcher
extern "C" void kernel_launch(void* const* d_in, const int* in_sizes, int n_in,
                              void* d_out, int out_size, void* d_ws, size_t ws_size,
                              hipStream_t stream) {
  const int*   edge_list  = (const int*)d_in[0];
  const int*   edge_type  = (const int*)d_in[1];
  const float* entity_emb = (const float*)d_in[2];
  const float* relemb     = (const float*)d_in[3];
  const float* W_src      = (const float*)d_in[4];
  const float* W_dst      = (const float*)d_in[5];
  const float* a_vec      = (const float*)d_in[6];
  const float* w_rel      = (const float*)d_in[7];
  const float* W_relproj  = (const float*)d_in[8];

  float* out_ent = (float*)d_out;                               // [40000,400]
  float* out_rel = out_ent + (size_t)N_NODES * OUT_W;           // [400,400]

  // workspace carve (floats): ~3.87 MB total
  float* ws     = (float*)d_ws;
  float* u      = ws;                    // 4*200
  float* v      = u + 4 * D_IN;          // 4*200
  float* dlog   = v + 4 * D_IN;          // 4*400
  float* mbuf   = dlog + 4 * N_REL;      // 4*400
  float* denom  = mbuf + 4 * N_REL;      // 4*400
  float* ebuf   = denom + 4 * N_REL;     // 4*100000
  float* agg    = ebuf + 4 * N_EDGES;    // 4*400*200
  float* relcat = agg + 4 * N_REL * D_IN;// 400*600 (16B-aligned offset)

  const dim3 wblk(32, 8);   // 8 waves / block for wave-per-item kernels

  k_init<<<(8 * N_REL + 4 * N_REL * D_IN + 255) / 256, 256, 0, stream>>>(mbuf, denom, agg);
  k_uv<<<(2 * 4 * D_IN + 255) / 256, 256, 0, stream>>>(W_src, W_dst, a_vec, u, v);
  k_dlogit<<<(4 * N_REL + 255) / 256, 256, 0, stream>>>(relemb, v, dlog);

  int ewaves = 4 * N_EDGES;
  k_edge_logits<<<(ewaves + 7) / 8, wblk, 0, stream>>>(edge_list, edge_type, entity_emb,
                                                       u, dlog, mbuf, ebuf);
  k_edge_exp<<<(4 * N_EDGES + 255) / 256, 256, 0, stream>>>(edge_type, mbuf, ebuf, denom);
  k_edge_agg<<<(ewaves + 7) / 8, wblk, 0, stream>>>(edge_list, edge_type, entity_emb,
                                                    ebuf, denom, agg);

  k_head_out<<<(N_REL * OUT_W + 7) / 8, wblk, 0, stream>>>(agg, W_src, relcat);
  k_cat_rel<<<(N_REL * D_IN + 255) / 256, 256, 0, stream>>>(relemb, relcat);

  // ent = entity_emb @ W_rel_proj : [40000,200]x[200,400]  (dominant GEMM)
  // M = 40000 = 625 * 64 -> MT=4, K=200, LDA=200, LDB=400, LDC=400.
  {
    int tiles = (N_NODES / 64) * (OUT_W / 16);   // 625*25 = 15625 waves
    k_wmma_gemm<4, 200, 200, 400, 400><<<(tiles + 3) / 4, dim3(32, 4), 0, stream>>>(
        entity_emb, W_relproj, out_ent, N_NODES, OUT_W);
  }
  // rel_final = relcat @ w_rel : [400,600]x[600,400]; MT=1, K=600.
  {
    int tiles = (N_REL / 16) * (OUT_W / 16);     // 25*25 = 625 waves
    k_wmma_gemm<1, 600, 600, 400, 400><<<(tiles + 3) / 4, dim3(32, 4), 0, stream>>>(
        relcat, w_rel, out_rel, N_REL, OUT_W);
  }

  (void)in_sizes; (void)n_in; (void)out_size; (void)ws_size;
}
